// ConvMambaLayer_50672024158838
// MI455X (gfx1250) — compile-verified, matrix operand order checked by reading, not run on hardware
//
#include <hip/hip_runtime.h>
#include <hip/hip_bf16.h>

typedef __attribute__((ext_vector_type(16))) _Float16 v16h;
typedef __attribute__((ext_vector_type(8)))  _Float16 v8h;
typedef __attribute__((ext_vector_type(8)))  float    v8f;
typedef __attribute__((ext_vector_type(4)))  float    v4f;

#define LTOT  16384   // 32*32*16 spatial positions
#define CHUNK 128     // scan chunk length
#define NCH   128     // number of chunks (CHUNK*NCH == LTOT)
#define GEMM_WAVES 2  // waves per GEMM block

__device__ __forceinline__ float sigf(float x)  { return 1.f / (1.f + __expf(-x)); }
__device__ __forceinline__ float siluf(float x) { return x * sigf(x); }
__device__ __forceinline__ float softplusf(float x) { return (x > 20.f) ? x : log1pf(__expf(x)); }

// ---------------------------------------------------------------------------
// Pack an f32 weight [Nw x K] into f16 [Npad x K], zero-padding rows >= Nw.
// Lets the GEMM load B fragments as unconditional b128 loads.
// ---------------------------------------------------------------------------
__global__ void k_packw(const float* __restrict__ W, _Float16* __restrict__ Wp,
                        int Nw, int K, int total) {
  int i = blockIdx.x * blockDim.x + threadIdx.x;
  if (i >= total) return;
  int n = i / K;
  Wp[i] = (n < Nw) ? (_Float16)W[i] : (_Float16)0.f;
}

// ---------------------------------------------------------------------------
// Prep: LayerNorm of mamba half (channels 32..63) -> xm [L x 32] row-major,
// and transpose conv half (channels 0..31) -> xcv [L x 32] row-major.
// ---------------------------------------------------------------------------
__global__ void k_prep(const float* __restrict__ x,
                       const float* __restrict__ ln_g, const float* __restrict__ ln_b,
                       float* __restrict__ xm, float* __restrict__ xcv) {
  int l = blockIdx.x * blockDim.x + threadIdx.x;
  if (l >= LTOT) return;
  float v[32];
  float s = 0.f;
  #pragma unroll
  for (int c = 0; c < 32; ++c) { v[c] = x[(32 + c) * LTOT + l]; s += v[c]; }
  float mu = s * (1.f / 32.f);
  float q = 0.f;
  #pragma unroll
  for (int c = 0; c < 32; ++c) { float d = v[c] - mu; q += d * d; }
  float inv = rsqrtf(q * (1.f / 32.f) + 1e-5f);
  #pragma unroll
  for (int c = 0; c < 32; ++c)
    xm[l * 32 + c] = (v[c] - mu) * inv * ln_g[c] + ln_b[c];
  #pragma unroll
  for (int c = 0; c < 32; ++c)
    xcv[l * 32 + c] = x[c * LTOT + l];
}

// ---------------------------------------------------------------------------
// WMMA GEMM:  C[M x ldc] = A[M x K] @ Wp^T (+bias) (+addC)
// A f32 row-major, Wp f16 pre-packed [Npad x K]. One 16x16 tile per wave.
// A tile staged to LDS (f32 -> f16, b128 traffic). K / bias / addC are
// compile-time so both the k-loop and the epilogue are straight-line code.
// ---------------------------------------------------------------------------
template<int K, bool HB, bool HA>
__global__ void k_gemm(const float* __restrict__ A, const _Float16* __restrict__ Wp,
                       const float* __restrict__ bias, const float* __restrict__ addC,
                       float* __restrict__ C, int ldc) {
  __shared__ _Float16 As[16 * 32];
  const int tid  = threadIdx.x;
  const int lane = tid & 31;
  const int wv   = tid >> 5;
  const int mt   = blockIdx.x;
  const int nt   = blockIdx.y * GEMM_WAVES + wv;
  const int mrow = lane & 15;
  const int hf   = lane >> 4;
  const int n    = nt * 16 + mrow;
  const int sr   = tid >> 2;        // staging row 0..15
  const int sq   = tid & 3;         // staging quarter 0..3

  // warm the A stream (global_prefetch on gfx1250)
  __builtin_prefetch(A + (mt * 16 + sr) * K, 0, 3);

  v8f acc = {};
  #pragma unroll
  for (int kt = 0; kt < K; kt += 32) {
    __syncthreads();
    {
      const float* src = A + (mt * 16 + sr) * K + kt + sq * 8;
      v4f f0 = *(const v4f*)(src);
      v4f f1 = *(const v4f*)(src + 4);
      v8h hh;
      hh[0] = (_Float16)f0[0]; hh[1] = (_Float16)f0[1];
      hh[2] = (_Float16)f0[2]; hh[3] = (_Float16)f0[3];
      hh[4] = (_Float16)f1[0]; hh[5] = (_Float16)f1[1];
      hh[6] = (_Float16)f1[2]; hh[7] = (_Float16)f1[3];
      *(v8h*)(&As[sr * 32 + sq * 8]) = hh;
    }
    __syncthreads();
    v8h a0 = *(const v8h*)(&As[mrow * 32 + hf * 8]);
    v8h a1 = *(const v8h*)(&As[mrow * 32 + 16 + hf * 8]);
    const _Float16* wrow = Wp + n * K + kt;
    v8h b0 = *(const v8h*)(wrow + hf * 8);
    v8h b1 = *(const v8h*)(wrow + 16 + hf * 8);
    v16h av = __builtin_shufflevector(a0, a1, 0,1,2,3,4,5,6,7,8,9,10,11,12,13,14,15);
    v16h bv = __builtin_shufflevector(b0, b1, 0,1,2,3,4,5,6,7,8,9,10,11,12,13,14,15);
    acc = __builtin_amdgcn_wmma_f32_16x16x32_f16(false, av, false, bv,
                                                 (short)0, acc, false, false);
  }
  if (nt * 16 < ldc) {               // uniform per-wave guard (padded tiles skip)
    float bval = HB ? bias[n] : 0.f;
    #pragma unroll
    for (int v = 0; v < 8; ++v) {
      int m = mt * 16 + hf * 8 + v;
      float rr = acc[v] + bval;
      if (HA) rr += addC[m * ldc + n];
      C[m * ldc + n] = rr;
    }
  }
}

// ---------------------------------------------------------------------------
// Depthwise causal conv1d (k=4) + SiLU over xi = xz[:, 0:64].
// rev=1 processes the time-reversed sequence (backward mamba branch).
// ---------------------------------------------------------------------------
__global__ void k_dwconv(const float* __restrict__ xz,
                         const float* __restrict__ w, const float* __restrict__ b,
                         float* __restrict__ xt, int rev) {
  int i = blockIdx.x * blockDim.x + threadIdx.x;
  if (i >= LTOT * 64) return;
  int l = i >> 6, d = i & 63;
  float acc = b[d];
  #pragma unroll
  for (int j = 0; j < 4; ++j) {
    int p = l - 3 + j;
    if (p >= 0) {
      int src = rev ? (LTOT - 1 - p) : p;
      acc += w[d * 4 + j] * xz[src * 128 + d];
    }
  }
  xt[l * 64 + d] = siluf(acc);
}

// ---------------------------------------------------------------------------
// Selective-scan pass 1: per-chunk local recurrence with h0=0.
// Grid (NCH, 32), block 32: two d-chains per wave (lanes 0-15 and 16-31),
// one lane per state n. Emits per-chunk decay product and final h.
// ---------------------------------------------------------------------------
__global__ void k_scan1(const float* __restrict__ xd, const float* __restrict__ u,
                        const float* __restrict__ Alog,
                        const float* __restrict__ dtw, const float* __restrict__ dtb,
                        float* __restrict__ prodA, float* __restrict__ hend) {
  int ch = blockIdx.x;
  int lane = threadIdx.x;
  int n = lane & 15;
  int d = blockIdx.y * 2 + (lane >> 4);
  float An = -__expf(Alog[d * 16 + n]);
  float w0 = dtw[d * 2], w1 = dtw[d * 2 + 1], bb = dtb[d];
  float h = 0.f, p = 1.f;
  int t0 = ch * CHUNK;
  for (int t = t0; t < t0 + CHUNK; ++t) {
    const float* row = xd + t * 48;
    float dl = softplusf(row[0] * w0 + row[1] * w1 + bb);
    float a  = __expf(dl * An);
    float uu = u[t * 64 + d];
    h = a * h + dl * row[2 + n] * uu;
    p *= a;
  }
  int idx = (ch * 64 + d) * 16 + n;
  prodA[idx] = p;
  hend[idx]  = h;
}

// Pass 2: exclusive scan of (prodA, hend) across chunks per (d, n) chain.
__global__ void k_scan2(const float* __restrict__ prodA, const float* __restrict__ hend,
                        float* __restrict__ hinit) {
  int i = blockIdx.x * blockDim.x + threadIdx.x;
  if (i >= 64 * 16) return;
  float carry = 0.f;
  for (int ch = 0; ch < NCH; ++ch) {
    int idx = ch * (64 * 16) + i;
    hinit[idx] = carry;
    carry = prodA[idx] * carry + hend[idx];
  }
}

// Pass 3: re-run local recurrence seeded with carry; reduce 16 states with
// width-16 wave shuffles; apply skip (u*D) and SiLU(z) gate; scatter.
// REV/ACCUM are compile-time so the inner loop has no uniform branches.
template<int REV, int ACCUM>
__global__ void k_scan3(const float* __restrict__ xd, const float* __restrict__ u,
                        const float* __restrict__ xz, const float* __restrict__ Alog,
                        const float* __restrict__ dtw, const float* __restrict__ dtb,
                        const float* __restrict__ Dp, const float* __restrict__ hinit,
                        float* __restrict__ ys) {
  int ch = blockIdx.x;
  int lane = threadIdx.x;
  int n = lane & 15;
  int d = blockIdx.y * 2 + (lane >> 4);
  float An = -__expf(Alog[d * 16 + n]);
  float w0 = dtw[d * 2], w1 = dtw[d * 2 + 1], bb = dtb[d], Dd = Dp[d];
  float h = hinit[(ch * 64 + d) * 16 + n];
  int t0 = ch * CHUNK;
  for (int t = t0; t < t0 + CHUNK; ++t) {
    const float* row = xd + t * 48;
    float dl = softplusf(row[0] * w0 + row[1] * w1 + bb);
    float a  = __expf(dl * An);
    float uu = u[t * 64 + d];
    h = a * h + dl * row[2 + n] * uu;
    float yp = h * row[18 + n];
    yp += __shfl_xor(yp, 1, 16);
    yp += __shfl_xor(yp, 2, 16);
    yp += __shfl_xor(yp, 4, 16);
    yp += __shfl_xor(yp, 8, 16);
    if (n == 0) {
      int lo = REV ? (LTOT - 1 - t) : t;
      float zv = xz[lo * 128 + 64 + d];
      float y  = (yp + uu * Dd) * siluf(zv);
      if (ACCUM) ys[lo * 64 + d] += y; else ys[lo * 64 + d] = y;
    }
  }
}

// ---------------------------------------------------------------------------
// Instance-norm statistics: one block per channel of buf [L x Cc].
// ---------------------------------------------------------------------------
__global__ void k_stats(const float* __restrict__ buf, float* __restrict__ stats, int Cc) {
  __shared__ float ss[256];
  __shared__ float sq[256];
  int c = blockIdx.x, tid = threadIdx.x;
  float s = 0.f, q = 0.f;
  for (int l = tid; l < LTOT; l += 256) {
    float v = buf[l * Cc + c];
    s += v; q += v * v;
  }
  ss[tid] = s; sq[tid] = q;
  __syncthreads();
  for (int st = 128; st > 0; st >>= 1) {
    if (tid < st) { ss[tid] += ss[tid + st]; sq[tid] += sq[tid + st]; }
    __syncthreads();
  }
  if (tid == 0) {
    float mu  = ss[0] * (1.f / (float)LTOT);
    float var = sq[0] * (1.f / (float)LTOT) - mu * mu;
    stats[2 * c]     = mu;
    stats[2 * c + 1] = rsqrtf(var + 1e-5f);
  }
}

__global__ void k_norm_lrelu(const float* __restrict__ in, float* __restrict__ out,
                             const float* __restrict__ stats, int Cc) {
  int i = blockIdx.x * blockDim.x + threadIdx.x;
  if (i >= LTOT * Cc) return;
  int c = i & (Cc - 1);   // Cc is a power of two (32)
  float v = (in[i] - stats[2 * c]) * stats[2 * c + 1];
  out[i] = (v >= 0.f) ? v : 0.1f * v;
}

// ---------------------------------------------------------------------------
// ELK stencils: 4 convs (3x3x3, 5x1x1, 1x5x1, 1x1x5), each 32->16 channels,
// concatenated into e4 [L x 64]. One block per voxel, one thread per out ch.
// ---------------------------------------------------------------------------
__global__ void k_elk(const float* __restrict__ h1,
                      const float* __restrict__ ew1, const float* __restrict__ eb1,
                      const float* __restrict__ ew2, const float* __restrict__ eb2,
                      const float* __restrict__ ew3, const float* __restrict__ eb3,
                      const float* __restrict__ ew4, const float* __restrict__ eb4,
                      float* __restrict__ e4) {
  const int l  = blockIdx.x;
  const int c  = threadIdx.x;           // 0..63
  const int g  = c >> 4, oc = c & 15;
  const int wx = l & 15, hy = (l >> 4) & 31, dz = l >> 9;
  float acc = 0.f;
  if (g == 0) {
    acc = eb1[oc];
    for (int kd = -1; kd <= 1; ++kd)
      for (int kh = -1; kh <= 1; ++kh)
        for (int kw = -1; kw <= 1; ++kw) {
          int zz = dz + kd, yy = hy + kh, xx = wx + kw;
          if ((unsigned)zz < 32u && (unsigned)yy < 32u && (unsigned)xx < 16u) {
            const float* row = h1 + ((zz * 32 + yy) * 16 + xx) * 32;
            const float* wp  = ew1 + (oc * 32) * 27 + (kd + 1) * 9 + (kh + 1) * 3 + (kw + 1);
            for (int ci = 0; ci < 32; ++ci) acc += wp[ci * 27] * row[ci];
          }
        }
  } else if (g == 1) {
    acc = eb2[oc];
    for (int kd = -2; kd <= 2; ++kd) {
      int zz = dz + kd;
      if ((unsigned)zz < 32u) {
        const float* row = h1 + ((zz * 32 + hy) * 16 + wx) * 32;
        const float* wp  = ew2 + (oc * 32) * 5 + (kd + 2);
        for (int ci = 0; ci < 32; ++ci) acc += wp[ci * 5] * row[ci];
      }
    }
  } else if (g == 2) {
    acc = eb3[oc];
    for (int kh = -2; kh <= 2; ++kh) {
      int yy = hy + kh;
      if ((unsigned)yy < 32u) {
        const float* row = h1 + ((dz * 32 + yy) * 16 + wx) * 32;
        const float* wp  = ew3 + (oc * 32) * 5 + (kh + 2);
        for (int ci = 0; ci < 32; ++ci) acc += wp[ci * 5] * row[ci];
      }
    }
  } else {
    acc = eb4[oc];
    for (int kw = -2; kw <= 2; ++kw) {
      int xx = wx + kw;
      if ((unsigned)xx < 16u) {
        const float* row = h1 + ((dz * 32 + hy) * 16 + xx) * 32;
        const float* wp  = ew4 + (oc * 32) * 5 + (kw + 2);
        for (int ci = 0; ci < 32; ++ci) acc += wp[ci * 5] * row[ci];
      }
    }
  }
  e4[l * 64 + c] = acc;
}

// ---------------------------------------------------------------------------
// Fusion kernels.
// ---------------------------------------------------------------------------
__global__ void k_pool(const float* __restrict__ h3, const float* __restrict__ xmam,
                       float* __restrict__ pooled) {
  __shared__ float ss[256];
  int c = blockIdx.x, tid = threadIdx.x;
  const float* src = (c & 1) ? (xmam + (c >> 1)) : (h3 + (c >> 1));
  float s = 0.f;
  for (int l = tid; l < LTOT; l += 256) s += src[l * 32];
  ss[tid] = s;
  __syncthreads();
  for (int st = 128; st > 0; st >>= 1) {
    if (tid < st) ss[tid] += ss[tid + st];
    __syncthreads();
  }
  if (tid == 0) pooled[c] = ss[0] * (1.f / (float)LTOT);
}

__global__ void k_attn(const float* __restrict__ atten_w, const float* __restrict__ pooled,
                       float* __restrict__ attn) {
  int c = threadIdx.x;
  float s = 0.f;
  for (int k = 0; k < 64; ++k) s += atten_w[c * 64 + k] * pooled[k];
  attn[c] = sigf(sigf(s));
}

__global__ void k_final(const float* __restrict__ h3, const float* __restrict__ xmam,
                        const float* __restrict__ attn,
                        const float* __restrict__ conv1_w,
                        const float* __restrict__ conv2_w, const float* __restrict__ conv2_b,
                        float* __restrict__ out) {
  __shared__ float fs[64];
  __shared__ float a2s;
  int l = blockIdx.x, c = threadIdx.x;
  float fc = (c & 1) ? xmam[l * 32 + (c >> 1)] : h3[l * 32 + (c >> 1)];
  fs[c] = fc * attn[c];
  __syncthreads();
  if (c == 0) {
    // attn2 uses the *concatenated* (not interleaved) tensor: [h3 | xmam]
    float s = conv2_b[0];
    for (int k = 0; k < 32; ++k) s += conv2_w[k]      * h3[l * 32 + k];
    for (int k = 0; k < 32; ++k) s += conv2_w[32 + k] * xmam[l * 32 + k];
    a2s = sigf(s);
  }
  __syncthreads();
  float s = 0.f;
  for (int k = 0; k < 64; ++k) s += conv1_w[c * 64 + k] * fs[k];
  out[c * LTOT + l] = s * a2s;
}

// ---------------------------------------------------------------------------
// Host-side launch
// ---------------------------------------------------------------------------
static float* wsfloat(void* ws, size_t& off, size_t n) {
  float* p = (float*)ws + off;
  off += n;
  return p;
}

extern "C" void kernel_launch(void* const* d_in, const int* in_sizes, int n_in,
                              void* d_out, int out_size, void* d_ws, size_t ws_size,
                              hipStream_t stream) {
  const float* x            = (const float*)d_in[0];
  const float* ln_g         = (const float*)d_in[1];
  const float* ln_b         = (const float*)d_in[2];
  const float* in_proj_w    = (const float*)d_in[3];
  const float* conv1d_w     = (const float*)d_in[4];
  const float* conv1d_bias  = (const float*)d_in[5];
  const float* x_proj_w     = (const float*)d_in[6];
  const float* dt_proj_w    = (const float*)d_in[7];
  const float* dt_proj_bias = (const float*)d_in[8];
  const float* A_log        = (const float*)d_in[9];
  const float* D_f          = (const float*)d_in[10];
  const float* conv1db_w    = (const float*)d_in[11];
  const float* conv1db_bias = (const float*)d_in[12];
  const float* x_projb_w    = (const float*)d_in[13];
  const float* dt_projb_w   = (const float*)d_in[14];
  const float* dt_projb_bias= (const float*)d_in[15];
  const float* Ab_log       = (const float*)d_in[16];
  const float* D_b          = (const float*)d_in[17];
  const float* out_proj_w   = (const float*)d_in[18];
  const float* cp1_w        = (const float*)d_in[19];
  const float* cp1_b        = (const float*)d_in[20];
  const float* elk1_w       = (const float*)d_in[21];
  const float* elk1_b       = (const float*)d_in[22];
  const float* elk2_w       = (const float*)d_in[23];
  const float* elk2_b       = (const float*)d_in[24];
  const float* elk3_w       = (const float*)d_in[25];
  const float* elk3_b       = (const float*)d_in[26];
  const float* elk4_w       = (const float*)d_in[27];
  const float* elk4_b       = (const float*)d_in[28];
  const float* elk5_w       = (const float*)d_in[29];
  const float* elk5_b       = (const float*)d_in[30];
  const float* cp2_w        = (const float*)d_in[31];
  const float* cp2_b        = (const float*)d_in[32];
  const float* atten_w      = (const float*)d_in[33];
  const float* conv1_w      = (const float*)d_in[34];
  const float* conv2_w      = (const float*)d_in[35];
  const float* conv2_b      = (const float*)d_in[36];
  float* out = (float*)d_out;

  size_t off = 0;
  float* xm    = wsfloat(d_ws, off, (size_t)LTOT * 32);
  float* xz    = wsfloat(d_ws, off, (size_t)LTOT * 128);
  float* xt_f  = wsfloat(d_ws, off, (size_t)LTOT * 64);
  float* xt_b  = wsfloat(d_ws, off, (size_t)LTOT * 64);
  float* xd_f  = wsfloat(d_ws, off, (size_t)LTOT * 48);
  float* xd_b  = wsfloat(d_ws, off, (size_t)LTOT * 48);
  float* prodA = wsfloat(d_ws, off, (size_t)NCH * 64 * 16);
  float* hend  = wsfloat(d_ws, off, (size_t)NCH * 64 * 16);
  float* hinit = wsfloat(d_ws, off, (size_t)NCH * 64 * 16);
  float* ysum  = wsfloat(d_ws, off, (size_t)LTOT * 64);
  float* xmam  = wsfloat(d_ws, off, (size_t)LTOT * 32);
  float* xcv   = wsfloat(d_ws, off, (size_t)LTOT * 32);
  float* t1    = wsfloat(d_ws, off, (size_t)LTOT * 32);
  float* h1    = wsfloat(d_ws, off, (size_t)LTOT * 32);
  float* e4    = wsfloat(d_ws, off, (size_t)LTOT * 64);
  float* h2    = wsfloat(d_ws, off, (size_t)LTOT * 32);
  float* h3    = wsfloat(d_ws, off, (size_t)LTOT * 32);
  float* stats = wsfloat(d_ws, off, 128);
  float* pooled= wsfloat(d_ws, off, 64);
  float* attn  = wsfloat(d_ws, off, 64);
  // packed f16 weights (carved from a float region; sizes keep 16B alignment)
  float* packbase = wsfloat(d_ws, off, 16384);
  _Float16* hp = (_Float16*)packbase;
  size_t ho = 0;
  _Float16* wp_in   = hp + ho; ho += 128 * 32;  // in_proj   Npad=128 K=32
  _Float16* wp_xf   = hp + ho; ho += 64 * 64;   // x_proj f  Npad=64  K=64
  _Float16* wp_xb   = hp + ho; ho += 64 * 64;   // x_proj b
  _Float16* wp_out  = hp + ho; ho += 32 * 64;   // out_proj  Npad=32  K=64
  _Float16* wp_cp1  = hp + ho; ho += 32 * 32;   // cp1
  _Float16* wp_elk5 = hp + ho; ho += 32 * 64;   // elk5
  _Float16* wp_cp2  = hp + ho; ho += 32 * 32;   // cp2
  (void)ws_size; (void)n_in; (void)in_sizes; (void)out_size;

  const int GB = GEMM_WAVES * 32;

  // --- one-shot weight packing (f32 -> zero-padded f16) ---
  k_packw<<<dim3(16), dim3(256), 0, stream>>>(in_proj_w, wp_in,  128, 32, 128 * 32);
  k_packw<<<dim3(16), dim3(256), 0, stream>>>(x_proj_w,  wp_xf,   34, 64, 64 * 64);
  k_packw<<<dim3(16), dim3(256), 0, stream>>>(x_projb_w, wp_xb,   34, 64, 64 * 64);
  k_packw<<<dim3(8),  dim3(256), 0, stream>>>(out_proj_w, wp_out, 32, 64, 32 * 64);
  k_packw<<<dim3(4),  dim3(256), 0, stream>>>(cp1_w,  wp_cp1,  32, 32, 32 * 32);
  k_packw<<<dim3(8),  dim3(256), 0, stream>>>(elk5_w, wp_elk5, 32, 64, 32 * 64);
  k_packw<<<dim3(4),  dim3(256), 0, stream>>>(cp2_w,  wp_cp2,  32, 32, 32 * 32);

  // --- prep (layernorm + transpose) ---
  k_prep<<<dim3(LTOT / 256), dim3(256), 0, stream>>>(x, ln_g, ln_b, xm, xcv);

  // --- mamba: in_proj GEMM [L,32]x[32,128] -> xz ---
  k_gemm<32, false, false><<<dim3(LTOT / 16, 4), dim3(GB), 0, stream>>>(
      xm, wp_in, nullptr, nullptr, xz, 128);

  // --- depthwise conv1d + silu, both directions ---
  k_dwconv<<<dim3((LTOT * 64) / 256), dim3(256), 0, stream>>>(xz, conv1d_w,  conv1d_bias,  xt_f, 0);
  k_dwconv<<<dim3((LTOT * 64) / 256), dim3(256), 0, stream>>>(xz, conv1db_w, conv1db_bias, xt_b, 1);

  // --- x_proj GEMMs [L,64]x[64,34] (ldc padded to 48; tile 3 store-skipped) ---
  k_gemm<64, false, false><<<dim3(LTOT / 16, 2), dim3(GB), 0, stream>>>(
      xt_f, wp_xf, nullptr, nullptr, xd_f, 48);
  k_gemm<64, false, false><<<dim3(LTOT / 16, 2), dim3(GB), 0, stream>>>(
      xt_b, wp_xb, nullptr, nullptr, xd_b, 48);

  // --- chunked selective scan, forward branch ---
  k_scan1<<<dim3(NCH, 32), dim3(32), 0, stream>>>(xd_f, xt_f, A_log, dt_proj_w, dt_proj_bias, prodA, hend);
  k_scan2<<<dim3(4), dim3(256), 0, stream>>>(prodA, hend, hinit);
  k_scan3<0, 0><<<dim3(NCH, 32), dim3(32), 0, stream>>>(xd_f, xt_f, xz, A_log, dt_proj_w, dt_proj_bias,
                                                        D_f, hinit, ysum);
  // --- backward branch (accumulates into ysum at original positions) ---
  k_scan1<<<dim3(NCH, 32), dim3(32), 0, stream>>>(xd_b, xt_b, Ab_log, dt_projb_w, dt_projb_bias, prodA, hend);
  k_scan2<<<dim3(4), dim3(256), 0, stream>>>(prodA, hend, hinit);
  k_scan3<1, 1><<<dim3(NCH, 32), dim3(32), 0, stream>>>(xd_b, xt_b, xz, Ab_log, dt_projb_w, dt_projb_bias,
                                                        D_b, hinit, ysum);

  // --- out_proj GEMM [L,64]x[64,32] -> x_mamba ---
  k_gemm<64, false, false><<<dim3(LTOT / 16, 1), dim3(GB), 0, stream>>>(
      ysum, wp_out, nullptr, nullptr, xmam, 32);

  // --- conv branch: cp1 (1x1x1) + instnorm + lrelu ---
  k_gemm<32, true, false><<<dim3(LTOT / 16, 1), dim3(GB), 0, stream>>>(
      xcv, wp_cp1, cp1_b, nullptr, t1, 32);
  k_stats<<<dim3(32), dim3(256), 0, stream>>>(t1, stats, 32);
  k_norm_lrelu<<<dim3((LTOT * 32) / 256), dim3(256), 0, stream>>>(t1, h1, stats, 32);

  // --- ELK stencils -> e4, elk5 (1x1x1 64->32) + residual, instnorm, lrelu ---
  k_elk<<<dim3(LTOT), dim3(64), 0, stream>>>(h1, elk1_w, elk1_b, elk2_w, elk2_b,
                                             elk3_w, elk3_b, elk4_w, elk4_b, e4);
  k_gemm<64, true, true><<<dim3(LTOT / 16, 1), dim3(GB), 0, stream>>>(
      e4, wp_elk5, elk5_b, h1, t1, 32);
  k_stats<<<dim3(32), dim3(256), 0, stream>>>(t1, stats, 32);
  k_norm_lrelu<<<dim3((LTOT * 32) / 256), dim3(256), 0, stream>>>(t1, h2, stats, 32);

  // --- cp2 (1x1x1) + instnorm + lrelu ---
  k_gemm<32, true, false><<<dim3(LTOT / 16, 1), dim3(GB), 0, stream>>>(
      h2, wp_cp2, cp2_b, nullptr, t1, 32);
  k_stats<<<dim3(32), dim3(256), 0, stream>>>(t1, stats, 32);
  k_norm_lrelu<<<dim3((LTOT * 32) / 256), dim3(256), 0, stream>>>(t1, h3, stats, 32);

  // --- fusion ---
  k_pool<<<dim3(64), dim3(256), 0, stream>>>(h3, xmam, pooled);
  k_attn<<<dim3(1), dim3(64), 0, stream>>>(atten_w, pooled, attn);
  k_final<<<dim3(LTOT), dim3(64), 0, stream>>>(h3, xmam, attn, conv1_w, conv2_w, conv2_b, out);
}
